// GraphRNNClassifier_77429670412620
// MI455X (gfx1250) — matrix operand order repaired and checked by reading, not compiled, not run on hardware
//
#include <hip/hip_runtime.h>
#include <hip/hip_bf16.h>
#include <math.h>

typedef __attribute__((ext_vector_type(16))) __bf16 v16bf;
typedef __attribute__((ext_vector_type(8)))  float  v8f;

constexpr int kB = 8, kT = 16, kN = 10000, kF = 64, kH = 64, kE = 1000000;
constexpr int kM = kB * kN;          // 80000 nodes per timestep
constexpr int kCls = 3;

// ------------------------------------------------------------------
// xw[m,h] = X[b,t,n,:] @ W_gcn   via bf16 WMMA (fp32 accumulate).
// One wave -> one 16-row x 64-col tile. 8 x v_wmma_f32_16x16x32_bf16.
// ------------------------------------------------------------------
__global__ void __launch_bounds__(256)
gemm_xw_kernel(const float* __restrict__ X, const float* __restrict__ Wg,
               float* __restrict__ xw, int t) {
  __shared__ __align__(32) __bf16 Wt[64 * 64];   // transposed: Wt[h][f], bf16
  int tid = threadIdx.x;
  // stage W_gcn^T into LDS as bf16 (8 KB)
  for (int i = 0; i < 16; ++i) {
    int idx = tid * 16 + i;
    int h = idx >> 6, f = idx & 63;
    Wt[idx] = (__bf16)Wg[f * 64 + h];
  }
  __syncthreads();

  const int wave = tid >> 5, lane = tid & 31;
  const int mrow = lane & 15, kg = lane >> 4;    // kg = K-half group (contiguous k order)
  const int base = (blockIdx.x * 8 + wave) * 16; // 16-row output tile
  const int r = base + mrow;
  const int b = r / kN, n = r - b * kN;
  const float* rowp = X + ((size_t)(b * kT + t) * kN + n) * kF;

  // A operands for K=0..31 (a0) and K=32..63 (a1), in contiguous k order (same as B):
  // vector elem j   -> K = kk + kg*16 + j
  // vector elem 8+j -> K = kk + kg*16 + 8 + j
  v16bf a0, a1;
  const int o = kg * 16;
#pragma unroll
  for (int j = 0; j < 8; ++j) {
    a0[j]     = (__bf16)rowp[o + j];
    a0[8 + j] = (__bf16)rowp[8 + o + j];
    a1[j]     = (__bf16)rowp[32 + o + j];
    a1[8 + j] = (__bf16)rowp[40 + o + j];
  }

  const int ncol = lane & 15;
#pragma unroll
  for (int nt = 0; nt < 4; ++nt) {
    // B operand: lane holds column (nt*16+ncol), elems j -> K = kk + kg*16 + j.
    // Wt is [col][k] so this is one contiguous 32B-aligned LDS vector load.
    v16bf b0 = *(const v16bf*)&Wt[(nt * 16 + ncol) * 64 + kg * 16];
    v16bf b1 = *(const v16bf*)&Wt[(nt * 16 + ncol) * 64 + 32 + kg * 16];
    v8f acc = {};
    acc = __builtin_amdgcn_wmma_f32_16x16x32_bf16(false, a0, false, b0,
                                                  (short)0, acc, false, false);
    acc = __builtin_amdgcn_wmma_f32_16x16x32_bf16(false, a1, false, b1,
                                                  (short)0, acc, false, false);
    // C/D layout: VGPR v -> row M = v + 8*kg, col N = lane&15
#pragma unroll
    for (int v = 0; v < 8; ++v) {
      int row = base + v + 8 * kg;
      xw[(size_t)row * 64 + nt * 16 + ncol] = acc[v];
    }
  }
}

// ------------------------------------------------------------------
// Per-timestep scratch init: agg = 0, deg = 1 (self loop)
// ------------------------------------------------------------------
__global__ void init_kernel(float4* __restrict__ agg4, float* __restrict__ deg) {
  int i = blockIdx.x * blockDim.x + threadIdx.x;
  if (i < kM * kH / 4) agg4[i] = make_float4(0.f, 0.f, 0.f, 0.f);
  if (i < kM) deg[i] = 1.0f;
}

__global__ void zero_psum_kernel(float* __restrict__ psum) {
  int i = blockIdx.x * blockDim.x + threadIdx.x;
  if (i < kT * kB * kH) psum[i] = 0.f;
}

// degree histogram over destinations (L2-resident fp32 atomics)
__global__ void deg_kernel(const int* __restrict__ EI, float* __restrict__ deg, int t) {
  int e = blockIdx.x * blockDim.x + threadIdx.x;
  if (e >= kE) return;
  int dst = EI[(size_t)t * 2 * kE + kE + e];
  unsafeAtomicAdd(&deg[dst], 1.0f);
}

__global__ void rsqrt_kernel(float* __restrict__ deg) {
  int m = blockIdx.x * blockDim.x + threadIdx.x;
  if (m < kM) deg[m] = rsqrtf(deg[m]);   // deg now holds dinv
}

// ------------------------------------------------------------------
// Edge scatter: 64 threads per edge (one channel each).
// Gathers hit L2 (20.5 MB xw working set << 192 MB L2).
// ------------------------------------------------------------------
__global__ void __launch_bounds__(256)
edge_kernel(const int* __restrict__ EI, const float* __restrict__ xw,
            const float* __restrict__ dinv, float* __restrict__ agg, int t) {
  long long gid = (long long)blockIdx.x * blockDim.x + threadIdx.x;
  int e = (int)(gid >> 6);
  int ch = (int)(gid & 63);
  if (e >= kE) return;
  const int* eit = EI + (size_t)t * 2 * kE;
  int src = eit[e];
  int dst = eit[kE + e];
  float coef = dinv[src] * dinv[dst];
  float v = xw[(size_t)src * 64 + ch] * coef;
  unsafeAtomicAdd(&agg[(size_t)dst * 64 + ch], v);
}

// ------------------------------------------------------------------
// pooled_sum[t,b,h] += sum_n (agg[m,h] + xw[m,h]*dinv[m]^2)
// (b_gcn and the 1/N mean are folded in later in the LSTM kernel)
// ------------------------------------------------------------------
__global__ void __launch_bounds__(256)
pool_kernel(const float* __restrict__ agg, const float* __restrict__ xw,
            const float* __restrict__ dinv, float* __restrict__ psum, int t) {
  int b = blockIdx.x / 20;
  int chunk = blockIdx.x % 20;         // 500 nodes per block
  int h = threadIdx.x & 63;
  int sub = threadIdx.x >> 6;          // 4 node-lanes per channel
  int n0 = chunk * 500;
  float acc = 0.f;
  for (int n = sub; n < 500; n += 4) {
    int m = b * kN + n0 + n;
    float di = dinv[m];
    acc += agg[(size_t)m * 64 + h] + xw[(size_t)m * 64 + h] * di * di;
  }
  unsafeAtomicAdd(&psum[t * (kB * kH) + b * kH + h], acc);
}

// ------------------------------------------------------------------
// LSTM over T steps + final FC. One workgroup; thread = (batch, unit).
// ------------------------------------------------------------------
__global__ void __launch_bounds__(512)
lstm_fc_kernel(const float* __restrict__ psum, const float* __restrict__ b_gcn,
               const float* __restrict__ W_ih, const float* __restrict__ W_hh,
               const float* __restrict__ b_ih, const float* __restrict__ b_hh,
               const float* __restrict__ W_fc, const float* __restrict__ b_fc,
               float* __restrict__ out) {
  __shared__ float xs[kB * kH], hs[kB * kH], cs[kB * kH];
  int tid = threadIdx.x;               // 512 threads: b = tid>>6, u = tid&63
  int b = tid >> 6, u = tid & 63;
  hs[tid] = 0.f;
  cs[tid] = 0.f;
  float bi = b_ih[u]       + b_hh[u];
  float bf = b_ih[64 + u]  + b_hh[64 + u];
  float bg = b_ih[128 + u] + b_hh[128 + u];
  float bo = b_ih[192 + u] + b_hh[192 + u];
  const float invN = 1.0f / (float)kN;

  for (int t = 0; t < kT; ++t) {
    xs[tid] = psum[t * (kB * kH) + tid] * invN + b_gcn[u];
    __syncthreads();                   // also orders previous h/c writes
    float zi = bi, zf = bf, zg = bg, zo = bo;
    const float* xb = &xs[b * 64];
    const float* hb = &hs[b * 64];
#pragma unroll 4
    for (int k = 0; k < 64; ++k) {
      float xv = xb[k], hv = hb[k];
      zi += W_ih[(size_t)u * 64 + k] * xv        + W_hh[(size_t)u * 64 + k] * hv;
      zf += W_ih[(size_t)(64 + u) * 64 + k] * xv + W_hh[(size_t)(64 + u) * 64 + k] * hv;
      zg += W_ih[(size_t)(128 + u) * 64 + k] * xv + W_hh[(size_t)(128 + u) * 64 + k] * hv;
      zo += W_ih[(size_t)(192 + u) * 64 + k] * xv + W_hh[(size_t)(192 + u) * 64 + k] * hv;
    }
    __syncthreads();                   // all reads of hs done before rewrite
    float si = 1.f / (1.f + __expf(-zi));
    float sf = 1.f / (1.f + __expf(-zf));
    float so = 1.f / (1.f + __expf(-zo));
    float c = sf * cs[tid] + si * tanhf(zg);
    cs[tid] = c;
    hs[tid] = so * tanhf(c);
  }
  __syncthreads();
  if (tid < kB * kCls) {
    int ob = tid / kCls, cls = tid % kCls;
    float acc = b_fc[cls];
    for (int k = 0; k < 64; ++k) acc += hs[ob * 64 + k] * W_fc[cls * 64 + k];
    out[ob * kCls + cls] = acc;
  }
}

// ------------------------------------------------------------------
extern "C" void kernel_launch(void* const* d_in, const int* in_sizes, int n_in,
                              void* d_out, int out_size, void* d_ws, size_t ws_size,
                              hipStream_t stream) {
  (void)in_sizes; (void)n_in; (void)out_size; (void)ws_size;
  const float* graph = (const float*)d_in[0];   // [B,T,N,F]
  const int*   ei    = (const int*)d_in[1];     // [T,2,E]
  const float* Wg    = (const float*)d_in[2];   // [F,H]
  const float* bg    = (const float*)d_in[3];   // [H]
  const float* Wih   = (const float*)d_in[4];   // [4H,H]
  const float* Whh   = (const float*)d_in[5];   // [4H,H]
  const float* bih   = (const float*)d_in[6];   // [4H]
  const float* bhh   = (const float*)d_in[7];   // [4H]
  const float* Wfc   = (const float*)d_in[8];   // [3,H]
  const float* bfc   = (const float*)d_in[9];   // [3]
  float* out = (float*)d_out;                   // [B,3] fp32

  // workspace layout (~41.3 MB, stays L2-resident per timestep)
  float* xw   = (float*)d_ws;                    // M*H
  float* agg  = xw + (size_t)kM * kH;            // M*H
  float* deg  = agg + (size_t)kM * kH;           // M   (degree -> dinv)
  float* psum = deg + kM;                        // T*B*H pooled partial sums

  zero_psum_kernel<<<(kT * kB * kH + 255) / 256, 256, 0, stream>>>(psum);

  for (int t = 0; t < kT; ++t) {
    gemm_xw_kernel<<<kM / 128, 256, 0, stream>>>(graph, Wg, xw, t);
    init_kernel<<<(kM * kH / 4 + 255) / 256, 256, 0, stream>>>((float4*)agg, deg);
    deg_kernel<<<(kE + 255) / 256, 256, 0, stream>>>(ei, deg, t);
    rsqrt_kernel<<<(kM + 255) / 256, 256, 0, stream>>>(deg);
    edge_kernel<<<(int)(((long long)kE * 64 + 255) / 256), 256, 0, stream>>>(ei, xw, deg, agg, t);
    pool_kernel<<<kB * 20, 256, 0, stream>>>(agg, xw, deg, psum, t);
  }
  lstm_fc_kernel<<<1, 512, 0, stream>>>(psum, bg, Wih, Whh, bih, bhh, Wfc, bfc, out);
}
